// multimodal_segmentation_84937273246096
// MI455X (gfx1250) — matrix operand order repaired and checked by reading, not compiled, not run on hardware
//
#include <hip/hip_runtime.h>
#include <math.h>

typedef __attribute__((ext_vector_type(16))) _Float16 v16h;
typedef __attribute__((ext_vector_type(8)))  _Float16 v8h;
typedef __attribute__((ext_vector_type(8)))  float    v8f;

union AF { v16h v; v8h h[2]; };

#define WMMA_F16(a, b, c) \
    __builtin_amdgcn_wmma_f32_16x16x32_f16(false, (a), false, (b), (short)0, (c), false, false)

static __device__ __forceinline__ v16h ld16h(const _Float16* p) { return *(const v16h*)p; }
static __device__ __forceinline__ v8h  ld8h (const _Float16* p) { return *(const v8h*)p;  }

// ---- gfx1250 async global->LDS path (ASYNCcnt), with portable fallback ----
#if __has_builtin(__builtin_amdgcn_global_load_async_to_lds_b128) && \
    __has_builtin(__builtin_amdgcn_s_wait_asynccnt)
#define USE_ASYNC_LDS 1
#else
#define USE_ASYNC_LDS 0
#endif

// Builtin expects pointers to 16B int vectors: global (AS1) src, LDS (AS3) dst.
typedef int i32x4v __attribute__((__vector_size__(16)));
typedef __attribute__((address_space(1))) i32x4v* gptr128;
typedef __attribute__((address_space(3))) i32x4v* lptr128;

static __device__ __forceinline__ void g2l16(const _Float16* g, _Float16* l) {
#if USE_ASYNC_LDS
    __builtin_amdgcn_global_load_async_to_lds_b128((gptr128)g, (lptr128)l, 0, 0);
#else
    *(uint4*)l = *(const uint4*)g;
#endif
}
static __device__ __forceinline__ void asyncWait() {
#if USE_ASYNC_LDS
    __builtin_amdgcn_s_wait_asynccnt(0);
#endif
}

// ---------------------------------------------------------------------------
// Kernel A: QKV projection. C[16384 x 2304] = x[16384 x 768] @ w_qkv + b_qkv
// fp32 operands -> register-staged pipeline with float4 loads, f16 LDS tiles.
// Scatters into q/k/v f16 buffers laid out [B*nH=384][N=512][hd=64].
// ---------------------------------------------------------------------------
__global__ __launch_bounds__(128)
void qkv_gemm(const float* __restrict__ x, const float* __restrict__ w,
              const float* __restrict__ bias,
              _Float16* __restrict__ qbuf, _Float16* __restrict__ kbuf,
              _Float16* __restrict__ vbuf)
{
    __shared__ __align__(32) _Float16 lA[64 * 32];   // [row][k]
    __shared__ __align__(32) _Float16 lBt[64 * 32];  // [n][k] (transposed)

    const int t = threadIdx.x, lane = t & 31, wv = t >> 5;
    const int ln = lane & 15, half = lane >> 4;
    const int aoff = half * 8, boff = half * 16;
    const int m0 = blockIdx.x * 64, n0 = blockIdx.y * 64;

    float4 ra[4], rb[4];

    auto gld = [&](int k0) {
#pragma unroll
        for (int i = 0; i < 4; i++) {
            int idx = t + i * 128;
            int arow = idx >> 3, ac4 = (idx & 7) * 4;
            ra[i] = *(const float4*)&x[(long)(m0 + arow) * 768 + k0 + ac4];
            int bkk = idx >> 4, bn4 = (idx & 15) * 4;
            rb[i] = *(const float4*)&w[(long)(k0 + bkk) * 2304 + n0 + bn4];
        }
    };
    auto stlds = [&]() {
#pragma unroll
        for (int i = 0; i < 4; i++) {
            int idx = t + i * 128;
            int arow = idx >> 3, ac4 = (idx & 7) * 4;
            union { _Float16 h[4]; uint2 u; } pk;
            pk.h[0] = (_Float16)ra[i].x; pk.h[1] = (_Float16)ra[i].y;
            pk.h[2] = (_Float16)ra[i].z; pk.h[3] = (_Float16)ra[i].w;
            *(uint2*)&lA[arow * 32 + ac4] = pk.u;
            int bkk = idx >> 4, bn4 = (idx & 15) * 4;
            lBt[(bn4 + 0) * 32 + bkk] = (_Float16)rb[i].x;
            lBt[(bn4 + 1) * 32 + bkk] = (_Float16)rb[i].y;
            lBt[(bn4 + 2) * 32 + bkk] = (_Float16)rb[i].z;
            lBt[(bn4 + 3) * 32 + bkk] = (_Float16)rb[i].w;
        }
    };

    v8f acc[4] = {};
    gld(0);
    for (int k0 = 0; k0 < 768; k0 += 32) {
        __syncthreads();            // all waves done reading previous LDS tile
        stlds();
        __syncthreads();            // tile visible
        if (k0 + 32 < 768) gld(k0 + 32);   // overlap next global loads w/ WMMA

        // Preload all fragments into distinct regs -> one wait, 4 back-to-back
        // WMMAs into the matrix pipe.
        AF a;
        a.h[0] = ld8h(&lA[(wv * 16 + ln) * 32 + boff]);
        a.h[1] = ld8h(&lA[(wv * 16 + ln) * 32 + boff + 8]);
        v16h bf0 = ld16h(&lBt[(0 * 16 + ln) * 32 + boff]);
        v16h bf1 = ld16h(&lBt[(1 * 16 + ln) * 32 + boff]);
        v16h bf2 = ld16h(&lBt[(2 * 16 + ln) * 32 + boff]);
        v16h bf3 = ld16h(&lBt[(3 * 16 + ln) * 32 + boff]);
        acc[0] = WMMA_F16(a.v, bf0, acc[0]);
        acc[1] = WMMA_F16(a.v, bf1, acc[1]);
        acc[2] = WMMA_F16(a.v, bf2, acc[2]);
        acc[3] = WMMA_F16(a.v, bf3, acc[3]);
    }

    // Epilogue: s (q/k/v select) and head are block-uniform scalars.
    const int s = n0 / 768;
    const int h0 = (n0 - s * 768) >> 6;
    _Float16* __restrict__ dstbuf = (s == 0) ? qbuf : (s == 1) ? kbuf : vbuf;
#pragma unroll
    for (int nt = 0; nt < 4; nt++) {
#pragma unroll
        for (int r = 0; r < 8; r++) {
            int m = m0 + wv * 16 + r + 8 * half;
            int n = n0 + nt * 16 + ln;
            int b = m >> 9, tok = m & 511;
            long dst = (((long)(b * 12 + h0)) * 512 + tok) * 64 + nt * 16 + ln;
            dstbuf[dst] = (_Float16)(acc[nt][r] + bias[n]);
        }
    }
}

// ---------------------------------------------------------------------------
// Kernel B: fused flash attention + decomposed 3D relative-position bias.
// One workgroup = (head bh, 64-query block); wave wv owns 16 queries.
// q + k tiles arrive via async global->LDS (double-buffered k); V tile is
// register-staged and transposed into LDS for B-fragments. q A-fragments are
// loop-invariant and hoisted out of the key loop.
// ---------------------------------------------------------------------------
__global__ __launch_bounds__(128)
void attention(const _Float16* __restrict__ qbuf, const _Float16* __restrict__ kbuf,
               const _Float16* __restrict__ vbuf,
               const float* __restrict__ Rd, const float* __restrict__ Rh,
               const float* __restrict__ Rw, _Float16* __restrict__ aout)
{
    __shared__ __align__(32) _Float16 qLDS[64 * 64];      // [qi][c]
    __shared__ float biasLDS[64 * 24];                    // [qi][kd|kh|kw]
    __shared__ __align__(32) _Float16 kLDS[2][32 * 64];   // [key][c], dbl buf
    __shared__ __align__(32) _Float16 vTLDS[64 * 32];     // [e][key]
    __shared__ __align__(32) _Float16 pLDS[4 * 16 * 32];  // per-wave P scratch

    const int t = threadIdx.x, lane = t & 31, wv = t >> 5;
    const int ln = lane & 15, half = lane >> 4;
    const int aoff = half * 8, boff = half * 16;
    const int qblock = blockIdx.x, bh = blockIdx.y;
    const float scale = 0.125f;  // 64^-0.5

    uint4 rv[2];
    auto issueK = [&](int kt, int buf) {
#pragma unroll
        for (int i = 0; i < 2; i++) {
            int idx = t + i * 128;
            int key = idx >> 3, c8 = (idx & 7) * 8;
            g2l16(&kbuf[((long)bh * 512 + kt * 32 + key) * 64 + c8],
                  &kLDS[buf][key * 64 + c8]);
        }
    };
    auto gldV = [&](int kt) {
#pragma unroll
        for (int i = 0; i < 2; i++) {
            int idx = t + i * 128;
            int key = idx >> 3, c8 = (idx & 7) * 8;
            rv[i] = *(const uint4*)&vbuf[((long)bh * 512 + kt * 32 + key) * 64 + c8];
        }
    };
    auto stV = [&]() {
#pragma unroll
        for (int i = 0; i < 2; i++) {
            int idx = t + i * 128;
            int key = idx >> 3, c8 = (idx & 7) * 8;
            union { uint4 u; _Float16 h[8]; } pk;
            pk.u = rv[i];
#pragma unroll
            for (int j = 0; j < 8; j++) vTLDS[(c8 + j) * 32 + key] = pk.h[j];
        }
    };

    // Stage 64 queries (async), prefetch first k tile + V registers.
    const _Float16* qsrc = qbuf + ((long)bh * 512 + qblock * 64) * 64;
#pragma unroll
    for (int i = 0; i < 4; i++) {
        int idx = t + i * 128;
        g2l16(&qsrc[idx * 8], &qLDS[idx * 8]);
    }
    issueK(0, 0);
    gldV(0);
    asyncWait();
    __syncthreads();

    // Decomposed rel-pos bias: 64 queries x (8 kd + 8 kh + 8 kw) dot products.
    {
        int qi = t & 63, grp = t >> 6;
        int tg = qblock * 64 + qi;
        int qd = tg >> 6, qh = (tg >> 3) & 7, qw = tg & 7;
        for (int j2 = 0; j2 < 12; j2++) {
            int j = grp * 12 + j2;
            const float* row;
            if (j < 8)       row = Rd + (qd - j + 7) * 64;
            else if (j < 16) row = Rh + (qh - (j - 8) + 7) * 64;
            else             row = Rw + (qw - (j - 16) + 7) * 64;
            const float4* r4 = (const float4*)row;
            float s = 0.f;
#pragma unroll
            for (int c4 = 0; c4 < 16; c4++) {
                float4 rr = r4[c4];
                s += (float)qLDS[qi * 64 + c4 * 4 + 0] * rr.x
                   + (float)qLDS[qi * 64 + c4 * 4 + 1] * rr.y
                   + (float)qLDS[qi * 64 + c4 * 4 + 2] * rr.z
                   + (float)qLDS[qi * 64 + c4 * 4 + 3] * rr.w;
            }
            biasLDS[qi * 24 + j] = s;
        }
    }
    __syncthreads();

    // q A-fragments (cb = 0 and 32 of hd) are invariant across key tiles.
    AF aq0, aq1;
    aq0.h[0] = ld8h(&qLDS[(wv * 16 + ln) * 64 + 0 + boff]);
    aq0.h[1] = ld8h(&qLDS[(wv * 16 + ln) * 64 + 0 + boff + 8]);
    aq1.h[0] = ld8h(&qLDS[(wv * 16 + ln) * 64 + 32 + boff]);
    aq1.h[1] = ld8h(&qLDS[(wv * 16 + ln) * 64 + 32 + boff + 8]);

    float mrow[8], lrow[8];
    v8f o[4] = {};
#pragma unroll
    for (int r = 0; r < 8; r++) { mrow[r] = -1e30f; lrow[r] = 0.f; }

    int buf = 0;
    for (int kt = 0; kt < 16; kt++) {
        __syncthreads();             // prev iter reads of vTLDS / kLDS done
        stV();                       // V tile (registers -> transposed LDS)
        asyncWait();                 // k tile [buf] landed in LDS
        __syncthreads();             // tiles visible to all waves
        if (kt < 15) { issueK(kt + 1, buf ^ 1); gldV(kt + 1); }

        // S = q . k^T, two 16-key subtiles; preload all 4 k B-fragments,
        // then issue 4 WMMAs interleaved across the two accumulators.
        v16h kb00 = ld16h(&kLDS[buf][(0  + ln) * 64 + 0  + boff]);
        v16h kb01 = ld16h(&kLDS[buf][(0  + ln) * 64 + 32 + boff]);
        v16h kb10 = ld16h(&kLDS[buf][(16 + ln) * 64 + 0  + boff]);
        v16h kb11 = ld16h(&kLDS[buf][(16 + ln) * 64 + 32 + boff]);
        v8f s0 = {}, s1 = {};
        s0 = WMMA_F16(aq0.v, kb00, s0);
        s1 = WMMA_F16(aq0.v, kb10, s1);
        s0 = WMMA_F16(aq1.v, kb01, s0);
        s1 = WMMA_F16(aq1.v, kb11, s1);

        float sv[2][8];
#pragma unroll
        for (int r = 0; r < 8; r++) {
            int qi64 = wv * 16 + r + 8 * half;
            int keyg0 = kt * 32 + ln;
            int keyg1 = keyg0 + 16;
            int kd0 = keyg0 >> 6, kh0 = (keyg0 >> 3) & 7, kw0 = keyg0 & 7;
            int kd1 = keyg1 >> 6, kh1 = (keyg1 >> 3) & 7, kw1 = keyg1 & 7;
            sv[0][r] = s0[r] * scale + biasLDS[qi64 * 24 + kd0]
                     + biasLDS[qi64 * 24 + 8 + kh0] + biasLDS[qi64 * 24 + 16 + kw0];
            sv[1][r] = s1[r] * scale + biasLDS[qi64 * 24 + kd1]
                     + biasLDS[qi64 * 24 + 8 + kh1] + biasLDS[qi64 * 24 + 16 + kw1];
        }

        // Online softmax: rows live across one 16-lane half -> xor 1,2,4,8.
#pragma unroll
        for (int r = 0; r < 8; r++) {
            float tmax = fmaxf(sv[0][r], sv[1][r]);
#pragma unroll
            for (int xm = 1; xm < 16; xm <<= 1)
                tmax = fmaxf(tmax, __shfl_xor(tmax, xm, 32));
            float mnew = fmaxf(mrow[r], tmax);
            float corr = __expf(mrow[r] - mnew);
            float p0 = __expf(sv[0][r] - mnew);
            float p1 = __expf(sv[1][r] - mnew);
            float rs = p0 + p1;
#pragma unroll
            for (int xm = 1; xm < 16; xm <<= 1)
                rs += __shfl_xor(rs, xm, 32);
            lrow[r] = lrow[r] * corr + rs;
            mrow[r] = mnew;
#pragma unroll
            for (int nt = 0; nt < 4; nt++) o[nt][r] *= corr;
            int qrow = r + 8 * half;
            pLDS[wv * 512 + qrow * 32 + ln]      = (_Float16)p0;
            pLDS[wv * 512 + qrow * 32 + 16 + ln] = (_Float16)p1;
        }
        // In-wave LDS RAW: DS ops are in-order per wave; flush and pin order.
        __builtin_amdgcn_wave_barrier();
        asm volatile("s_wait_dscnt 0" ::: "memory");

        AF pa;
        pa.h[0] = ld8h(&pLDS[wv * 512 + ln * 32 + boff]);
        pa.h[1] = ld8h(&pLDS[wv * 512 + ln * 32 + boff + 8]);
        v16h vf0 = ld16h(&vTLDS[(0 * 16 + ln) * 32 + boff]);
        v16h vf1 = ld16h(&vTLDS[(1 * 16 + ln) * 32 + boff]);
        v16h vf2 = ld16h(&vTLDS[(2 * 16 + ln) * 32 + boff]);
        v16h vf3 = ld16h(&vTLDS[(3 * 16 + ln) * 32 + boff]);
        o[0] = WMMA_F16(pa.v, vf0, o[0]);
        o[1] = WMMA_F16(pa.v, vf1, o[1]);
        o[2] = WMMA_F16(pa.v, vf2, o[2]);
        o[3] = WMMA_F16(pa.v, vf3, o[3]);
        buf ^= 1;
    }

    // Normalize and write f16 attention output [b][tok][h*64+e].
    int b = bh / 12, hh = bh % 12;
#pragma unroll
    for (int nt = 0; nt < 4; nt++) {
#pragma unroll
        for (int r = 0; r < 8; r++) {
            int tok = qblock * 64 + wv * 16 + r + 8 * half;
            float inv = 1.0f / lrow[r];
            aout[((long)(b * 512 + tok)) * 768 + hh * 64 + nt * 16 + ln] =
                (_Float16)(o[nt][r] * inv);
        }
    }
}

// ---------------------------------------------------------------------------
// Kernel C: output projection. out[16384 x 768] = aout(f16) @ w_proj + b_proj
// A tile is f16 -> async global->LDS double-buffered; B register-staged fp32.
// ---------------------------------------------------------------------------
__global__ __launch_bounds__(128)
void proj_gemm(const _Float16* __restrict__ aout, const float* __restrict__ w,
               const float* __restrict__ bias, float* __restrict__ out)
{
    __shared__ __align__(32) _Float16 lA2[2][64 * 32];
    __shared__ __align__(32) _Float16 lBt[64 * 32];

    const int t = threadIdx.x, lane = t & 31, wv = t >> 5;
    const int ln = lane & 15, half = lane >> 4;
    const int aoff = half * 8, boff = half * 16;
    const int m0 = blockIdx.x * 64, n0 = blockIdx.y * 64;

    float4 rb[4];
    auto issueA = [&](int k0, int bf) {
#pragma unroll
        for (int i = 0; i < 2; i++) {
            int idx = t + i * 128;
            int arow = idx >> 2, ac8 = (idx & 3) * 8;
            g2l16(&aout[(long)(m0 + arow) * 768 + k0 + ac8],
                  &lA2[bf][arow * 32 + ac8]);
        }
    };
    auto gldB = [&](int k0) {
#pragma unroll
        for (int i = 0; i < 4; i++) {
            int idx = t + i * 128;
            int bkk = idx >> 4, bn4 = (idx & 15) * 4;
            rb[i] = *(const float4*)&w[(long)(k0 + bkk) * 768 + n0 + bn4];
        }
    };
    auto stB = [&]() {
#pragma unroll
        for (int i = 0; i < 4; i++) {
            int idx = t + i * 128;
            int bkk = idx >> 4, bn4 = (idx & 15) * 4;
            lBt[(bn4 + 0) * 32 + bkk] = (_Float16)rb[i].x;
            lBt[(bn4 + 1) * 32 + bkk] = (_Float16)rb[i].y;
            lBt[(bn4 + 2) * 32 + bkk] = (_Float16)rb[i].z;
            lBt[(bn4 + 3) * 32 + bkk] = (_Float16)rb[i].w;
        }
    };

    v8f acc[4] = {};
    issueA(0, 0);
    gldB(0);
    int buf = 0;
    for (int k0 = 0; k0 < 768; k0 += 32) {
        __syncthreads();
        stB();
        asyncWait();
        __syncthreads();
        if (k0 + 32 < 768) { issueA(k0 + 32, buf ^ 1); gldB(k0 + 32); }

        AF a;
        a.h[0] = ld8h(&lA2[buf][(wv * 16 + ln) * 32 + boff]);
        a.h[1] = ld8h(&lA2[buf][(wv * 16 + ln) * 32 + boff + 8]);
        v16h bf0 = ld16h(&lBt[(0 * 16 + ln) * 32 + boff]);
        v16h bf1 = ld16h(&lBt[(1 * 16 + ln) * 32 + boff]);
        v16h bf2 = ld16h(&lBt[(2 * 16 + ln) * 32 + boff]);
        v16h bf3 = ld16h(&lBt[(3 * 16 + ln) * 32 + boff]);
        acc[0] = WMMA_F16(a.v, bf0, acc[0]);
        acc[1] = WMMA_F16(a.v, bf1, acc[1]);
        acc[2] = WMMA_F16(a.v, bf2, acc[2]);
        acc[3] = WMMA_F16(a.v, bf3, acc[3]);
        buf ^= 1;
    }

#pragma unroll
    for (int nt = 0; nt < 4; nt++) {
#pragma unroll
        for (int r = 0; r < 8; r++) {
            int m = m0 + wv * 16 + r + 8 * half;
            int n = n0 + nt * 16 + ln;
            out[(long)m * 768 + n] = acc[nt][r] + bias[n];
        }
    }
}

// ---------------------------------------------------------------------------
extern "C" void kernel_launch(void* const* d_in, const int* in_sizes, int n_in,
                              void* d_out, int out_size, void* d_ws, size_t ws_size,
                              hipStream_t stream)
{
    (void)in_sizes; (void)n_in; (void)out_size; (void)ws_size;
    const float* x      = (const float*)d_in[0];
    const float* w_qkv  = (const float*)d_in[1];
    const float* b_qkv  = (const float*)d_in[2];
    const float* w_proj = (const float*)d_in[3];
    const float* b_proj = (const float*)d_in[4];
    const float* Rd     = (const float*)d_in[5];
    const float* Rh     = (const float*)d_in[6];
    const float* Rw     = (const float*)d_in[7];
    float* out = (float*)d_out;

    _Float16* ws = (_Float16*)d_ws;
    const long SEG = 384L * 512 * 64;  // 12,582,912 f16 elements per buffer
    _Float16* qbuf = ws;
    _Float16* kbuf = ws + SEG;
    _Float16* vbuf = ws + 2 * SEG;
    _Float16* abuf = ws + 3 * SEG;

    // M=16384 (256 tiles of 64), N=2304 (36 tiles of 64)
    qkv_gemm<<<dim3(256, 36), 128, 0, stream>>>(x, w_qkv, b_qkv, qbuf, kbuf, vbuf);
    // 8 query blocks x 384 (batch*head)
    attention<<<dim3(8, 384), 128, 0, stream>>>(qbuf, kbuf, vbuf, Rd, Rh, Rw, abuf);
    // M=16384, N=768 (12 tiles of 64)
    proj_gemm<<<dim3(256, 12), 128, 0, stream>>>(abuf, w_proj, b_proj, out);
}
